// MultiheadAttention_67559835566643
// MI455X (gfx1250) — matrix operand order repaired and checked by reading, not compiled, not run on hardware
//
#include <hip/hip_runtime.h>

#define TSEQ 1024
#define BSZ  4
#define EMB  1024
#define NH   16
#define HD   64
#define NB   64          // BSZ*NH attention batches
#define ROWS 4096        // TSEQ*BSZ
#define SLEN 1025        // src len incl. bias-kv token
#define SPAD 1088        // 17 chunks of 64
#define PBW  2112        // padded rel-pos table width (covers s-t+1023 up to 2110)
#define SCALE_Q 0.00390625f  // (hd^-0.5)/ALPHA = (1/8)/32

typedef __attribute__((ext_vector_type(16))) _Float16 v16h;
typedef __attribute__((ext_vector_type(8)))  _Float16 v8h;
typedef __attribute__((ext_vector_type(8)))  float    v8f;
typedef __attribute__((ext_vector_type(4)))  float    v4f;

union H16 { v16h v; v8h h[2]; };

// ---- async global->LDS staging (CDNA5 GLOBAL_LOAD_ASYNC_TO_LDS_B128, ASYNCcnt) ----
#if __has_builtin(__builtin_amdgcn_global_load_async_to_lds_b128) && __has_builtin(__builtin_amdgcn_s_wait_asynccnt)
#define USE_ASYNC_LDS 1
typedef int v4i_g __attribute__((vector_size(16)));
typedef __attribute__((address_space(1))) v4i_g as1_v4i;
typedef __attribute__((address_space(3))) v4i_g as3_v4i;
#else
#define USE_ASYNC_LDS 0
#endif

__device__ __forceinline__ void stage_b128(const _Float16* g, _Float16* l) {
#if USE_ASYNC_LDS
    __builtin_amdgcn_global_load_async_to_lds_b128(
        (as1_v4i*)(unsigned long long)(uintptr_t)g,
        (as3_v4i*)(unsigned)(uintptr_t)l, 0, 0);
#else
    *(v8h*)l = *(const v8h*)g;
#endif
}

__device__ __forceinline__ void wait_async_lds() {
#if USE_ASYNC_LDS
    __builtin_amdgcn_s_wait_asynccnt(0);
#endif
}

// A-fragment (16x32 f16, M x K): lane M = lane&15; K groups depend on lane>>4.
// Lanes 0-15 hold K 0..15; lanes 16-31 hold K 16..31 (contiguous).
__device__ __forceinline__ v16h load_a_frag(const _Float16* base, int stride, int lane) {
    int M = lane & 15, g = (lane >> 4) & 1;
    const _Float16* p = base + M * stride + g * 16;
    H16 u;
    u.h[0] = *(const v8h*)(p);        // K = g*16 .. g*16+7
    u.h[1] = *(const v8h*)(p + 8);    // K = g*16+8 .. g*16+15
    return u.v;
}

// B-fragment (32x16 f16, K x N) loaded from a B^T ([n][k]) buffer:
// lane n = lane&15, K = (lane>>4)*16 + 0..15 contiguous -> one 32B load.
__device__ __forceinline__ v16h load_b_frag(const _Float16* baseT, int stride, int lane) {
    int n = lane & 15, g = (lane >> 4) & 1;
    return *(const v16h*)(baseT + n * stride + g * 16);
}

__device__ __forceinline__ v8f wmma_f16(v16h a, v16h b, v8f c) {
    return __builtin_amdgcn_wmma_f32_16x16x32_f16(false, a, false, b, (short)0, c, false, false);
}

// ---------------- setup kernels ----------------

__global__ void k_cvt_f16(const float* __restrict__ src, _Float16* __restrict__ dst, int n) {
    int i = (blockIdx.x * blockDim.x + threadIdx.x) * 4;
    if (i < n) {
        v4f x = *(const v4f*)(src + i);
        dst[i + 0] = (_Float16)x.x; dst[i + 1] = (_Float16)x.y;
        dst[i + 2] = (_Float16)x.z; dst[i + 3] = (_Float16)x.w;
    }
}

__global__ void k_transpose_w(const float* __restrict__ W, _Float16* __restrict__ Wt) {
    int idx = blockIdx.x * blockDim.x + threadIdx.x;   // EMB*EMB threads
    int k = idx >> 10, n = idx & 1023;
    Wt[(size_t)n * EMB + k] = (_Float16)W[(size_t)k * EMB + n];
}

// Fill bias-kv row (s=1024) and zero padding rows 1025..1087 of Kh and Vt.
__global__ void k_biasrows(const float* __restrict__ bias_k, const float* __restrict__ bias_v,
                           _Float16* __restrict__ Kh, _Float16* __restrict__ Vt) {
    int idx = blockIdx.x * blockDim.x + threadIdx.x;   // NB*HD*64
    int batch = idx >> 12;
    int d  = (idx >> 6) & 63;
    int sp = idx & 63;
    int h = batch & 15;
    int s = 1024 + sp;
    _Float16 vk = (sp == 0) ? (_Float16)bias_k[h * 64 + d] : (_Float16)0.f;
    _Float16 vv = (sp == 0) ? (_Float16)bias_v[h * 64 + d] : (_Float16)0.f;
    Kh[(size_t)batch * SPAD * HD + (size_t)s * HD + d] = vk;
    Vt[(size_t)batch * HD * SPAD + (size_t)d * SPAD + s] = vv;
}

// Per-head positional-bias table: pbh[h][p] = rel_emb[bucket(p-1023)][h], p in [0,PBW).
__global__ void k_pbh(const float* __restrict__ rel_emb, _Float16* __restrict__ pbh) {
    int idx = blockIdx.x * blockDim.x + threadIdx.x;   // NH*PBW
    if (idx >= NH * PBW) return;
    int h = idx / PBW;
    int p = idx - h * PBW;
    int pc = p > 2047 ? 2047 : p;                      // entries >2047 are masked anyway
    int rel = pc - 1023;
    int bkt = rel > 0 ? 16 : 0;
    int a = rel < 0 ? -rel : rel;
    int r;
    if (a < 8) r = a;
    else {
        int large = 8 + (int)(__logf((float)a / 8.0f + 1e-6f) / __logf(16.0f) * 8.0f);
        r = large < 15 ? large : 15;
    }
    pbh[idx] = (_Float16)rel_emb[(bkt + r) * NH + h];
}

// gwsum[d] = sum_j grep_w[d][j]; gwsum[64] = sum_j grep_b[j]
__global__ void k_gwsum(const float* __restrict__ grep_w, const float* __restrict__ grep_b,
                        float* __restrict__ gws) {
    int d = threadIdx.x;
    if (d < 64) {
        float s = 0.f;
        for (int j = 0; j < 8; ++j) s += grep_w[d * 8 + j];
        gws[d] = s;
    }
    if (d == 0) {
        float s = 0.f;
        for (int j = 0; j < 8; ++j) s += grep_b[j];
        gws[64] = s;
    }
}

__global__ void k_gate(const _Float16* __restrict__ Qh, const float* __restrict__ gws,
                       const float* __restrict__ grep_a, float* __restrict__ gate) {
    int idx = blockIdx.x * blockDim.x + threadIdx.x;   // NB*TSEQ
    int batch = idx >> 10, t = idx & 1023;
    int h = batch & 15;
    const _Float16* q = Qh + (size_t)batch * TSEQ * HD + (size_t)t * HD;
    float s = gws[64];
    for (int d = 0; d < 64; ++d) s += (float)q[d] * gws[d];
    float g = 1.0f / (1.0f + __expf(-s));
    gate[idx] = g * grep_a[h];
}

// ---------------- WMMA GEMM: C(4096x1024) = A(4096x1024) x W(1024x1024) + bias ----------------
// K-step 64, double-buffered LDS with async global->LDS staging; 16 WMMA per barrier.
// MODE 0: scatter to Qh[b*16+h][t][d]    MODE 1: Kh[b*16+h][s][d]
// MODE 2: Vt[b*16+h][d][s]               MODE 3: f32 out[row][col]
template<int MODE>
__global__ void __launch_bounds__(256)
k_gemm_wmma(const _Float16* __restrict__ Ag, const _Float16* __restrict__ Bg,
            const float* __restrict__ bias, void* __restrict__ outp) {
    __shared__ __align__(32) _Float16 As[2][128 * 64];   // [m][k]
    __shared__ __align__(32) _Float16 Bs[2][128 * 64];   // [n][k]  (B^T layout)
    int tid = threadIdx.x;
    int lane = tid & 31;
    int wv = tid >> 5;
    int wr = wv >> 2, wc = wv & 3;                       // 2x4 wave grid; wave = 64Mx32N
    int m0 = blockIdx.x * 128, n0 = blockIdx.y * 128;

    v8f zero = {};
    v8f acc[4][2];
    for (int i = 0; i < 4; ++i) for (int j = 0; j < 2; ++j) acc[i][j] = zero;

    auto stage = [&](int buf, int kt) {
        int k0 = kt * 64;
        for (int j = 0; j < 4; ++j) {
            int c = tid + j * 256;                       // 0..1023 chunks of 8 f16
            int row = c >> 3, kc = (c & 7) * 8;
            stage_b128(Ag + (size_t)(m0 + row) * EMB + k0 + kc, &As[buf][row * 64 + kc]);
            stage_b128(Bg + (size_t)(n0 + row) * EMB + k0 + kc, &Bs[buf][row * 64 + kc]);
        }
    };

    stage(0, 0);
    for (int kt = 0; kt < 16; ++kt) {
        int buf = kt & 1;
        wait_async_lds();
        __syncthreads();
        if (kt + 1 < 16) stage(buf ^ 1, kt + 1);
        if (kt + 2 < 16) {                               // L2 prefetch 2 stages ahead
            __builtin_prefetch(Ag + (size_t)(m0 + (tid >> 1)) * EMB + (kt + 2) * 64, 0, 1);
            __builtin_prefetch(Bg + (size_t)(n0 + (tid >> 1)) * EMB + (kt + 2) * 64, 0, 1);
        }
        for (int ks = 0; ks < 2; ++ks) {
            v16h af[4], bf[2];
            for (int mi = 0; mi < 4; ++mi)
                af[mi] = load_a_frag(&As[buf][(wr * 64 + mi * 16) * 64 + ks * 32], 64, lane);
            for (int ni = 0; ni < 2; ++ni)
                bf[ni] = load_b_frag(&Bs[buf][(wc * 32 + ni * 16) * 64 + ks * 32], 64, lane);
            for (int mi = 0; mi < 4; ++mi)
                for (int ni = 0; ni < 2; ++ni)
                    acc[mi][ni] = wmma_f16(af[mi], bf[ni], acc[mi][ni]);
        }
    }

    int g = (lane >> 4) & 1, ln = lane & 15;
    for (int mi = 0; mi < 4; ++mi) for (int ni = 0; ni < 2; ++ni) {
        for (int i = 0; i < 8; ++i) {
            int row = m0 + wr * 64 + mi * 16 + i + 8 * g;   // C layout: M = vgpr + 8*(lane>>4)
            int col = n0 + wc * 32 + ni * 16 + ln;          // N = lane&15
            float val = acc[mi][ni][i] + bias[col];
            if (MODE == 3) {
                ((float*)outp)[(size_t)row * EMB + col] = val;
            } else {
                int t = row >> 2, b = row & 3, h = col >> 6, d = col & 63;
                int batch = b * 16 + h;
                _Float16 hv = (_Float16)val;
                _Float16* O = (_Float16*)outp;
                if (MODE == 0)      O[(size_t)batch * (TSEQ * HD) + t * 64 + d] = hv;
                else if (MODE == 1) O[(size_t)batch * (SPAD * HD) + t * 64 + d] = hv;
                else                O[(size_t)batch * (SPAD * HD) + d * SPAD + t] = hv;
            }
        }
    }
}

// ---------------- flash attention with gated T5 bias ----------------
__global__ void __launch_bounds__(256)
k_attn(const _Float16* __restrict__ Qh, const _Float16* __restrict__ Kh,
       const _Float16* __restrict__ Vt, const float* __restrict__ gate,
       const _Float16* __restrict__ pbh, _Float16* __restrict__ A2) {
    __shared__ __align__(32) _Float16 Ks[2][64 * 64];     // [s_local][d]
    __shared__ __align__(32) _Float16 Vs[2][64 * 64];     // [d][s_local]  (V^T)
    __shared__ __align__(32) _Float16 Ps[8 * 16 * 64];    // per-wave probs [m][s_local]
    __shared__ _Float16 spb[PBW];                         // per-head f16 positional bias

    int tid = threadIdx.x;
    int lane = tid & 31, wv = tid >> 5;
    int g = (lane >> 4) & 1, ln = lane & 15;
    int batch = blockIdx.x;
    int h = batch & 15, b = batch >> 4;
    int t0 = blockIdx.y * 128 + wv * 16;                  // 16 rows per wave

    for (int j = 0; j < 9; ++j) {
        int i = tid + j * 256;
        if (i < PBW) spb[i] = pbh[h * PBW + i];
    }

    const _Float16* Qbase = Qh + (size_t)batch * TSEQ * HD + (size_t)t0 * HD;
    v16h qa0 = load_a_frag(Qbase, 64, lane);              // K(d) = 0..31
    v16h qa1 = load_a_frag(Qbase + 32, 64, lane);         // K(d) = 32..63

    float grow[8];
    for (int i = 0; i < 8; ++i) grow[i] = gate[batch * TSEQ + t0 + i + 8 * g];

    float m_i[8], l_i[8];
    for (int i = 0; i < 8; ++i) { m_i[i] = -1e30f; l_i[i] = 0.f; }
    v8f zero = {};
    v8f oacc[4];
    for (int dt = 0; dt < 4; ++dt) oacc[dt] = zero;

    const _Float16* Kb = Kh + (size_t)batch * SPAD * HD;
    const _Float16* Vb = Vt + (size_t)batch * HD * SPAD;
    _Float16* Pw = Ps + wv * 16 * 64;

    auto stage_chunk = [&](int buf, int sc) {
        int sbase = sc * 64;
        for (int j = 0; j < 2; ++j) {
            int c = tid + j * 256;                        // 0..511 chunks of 8 f16
            int r = c >> 3, cc = (c & 7) * 8;
            stage_b128(Kb + (size_t)(sbase + r) * HD + cc, &Ks[buf][r * 64 + cc]);
            stage_b128(Vb + (size_t)r * SPAD + sbase + cc, &Vs[buf][r * 64 + cc]);
        }
    };

    stage_chunk(0, 0);
    for (int sc = 0; sc < 17; ++sc) {
        int buf = sc & 1;
        int sbase = sc * 64;
        wait_async_lds();
        __syncthreads();
        if (sc + 1 < 17) stage_chunk(buf ^ 1, sc + 1);

        // scores: Q(16x64) @ K^T -> 4 tiles of 16x16, 2 WMMA each
        v8f sacc[4];
        for (int nt = 0; nt < 4; ++nt) sacc[nt] = zero;
        for (int nt = 0; nt < 4; ++nt) {
            v16h b0 = load_b_frag(&Ks[buf][(nt * 16) * 64], 64, lane);
            v16h b1 = load_b_frag(&Ks[buf][(nt * 16) * 64 + 32], 64, lane);
            sacc[nt] = wmma_f16(qa0, b0, sacc[nt]);
            sacc[nt] = wmma_f16(qa1, b1, sacc[nt]);
        }

        // scale + gated positional bias + mask (single f16 LDS lookup per element)
        for (int nt = 0; nt < 4; ++nt) {
            int s = sbase + nt * 16 + ln;
            for (int i = 0; i < 8; ++i) {
                int t = t0 + i + 8 * g;
                float pb = (float)spb[s - t + 1023];      // in [0, PBW) by construction
                float v = sacc[nt][i] * SCALE_Q + grow[i] * pb;
                sacc[nt][i] = (s < SLEN) ? v : -1e30f;
            }
        }

        // online softmax (rows live across 16-lane groups)
        float alpha[8], mnew[8];
        for (int i = 0; i < 8; ++i) {
            float cm = fmaxf(fmaxf(sacc[0][i], sacc[1][i]), fmaxf(sacc[2][i], sacc[3][i]));
            cm = fmaxf(cm, __shfl_xor(cm, 1, 32));
            cm = fmaxf(cm, __shfl_xor(cm, 2, 32));
            cm = fmaxf(cm, __shfl_xor(cm, 4, 32));
            cm = fmaxf(cm, __shfl_xor(cm, 8, 32));
            mnew[i] = fmaxf(m_i[i], cm);
            alpha[i] = __expf(m_i[i] - mnew[i]);
            m_i[i] = mnew[i];
        }
        for (int i = 0; i < 8; ++i) {
            float rs = 0.f;
            for (int nt = 0; nt < 4; ++nt) {
                float p = __expf(sacc[nt][i] - mnew[i]);
                rs += p;
                Pw[(i + 8 * g) * 64 + nt * 16 + ln] = (_Float16)p;   // C-layout -> [m][s]
            }
            rs += __shfl_xor(rs, 1, 32);
            rs += __shfl_xor(rs, 2, 32);
            rs += __shfl_xor(rs, 4, 32);
            rs += __shfl_xor(rs, 8, 32);
            l_i[i] = l_i[i] * alpha[i] + rs;
        }
        for (int dt = 0; dt < 4; ++dt)
            for (int i = 0; i < 8; ++i)
                oacc[dt][i] *= alpha[i];

        // wave-local LDS RAW fence before reloading probs as A-fragments
        asm volatile("s_wait_dscnt 0" ::: "memory");

        // O += P(16x64) @ V(64x64)
        for (int kk = 0; kk < 2; ++kk) {
            v16h pa = load_a_frag(Pw + kk * 32, 64, lane);
            for (int dt = 0; dt < 4; ++dt) {
                v16h vb = load_b_frag(&Vs[buf][(dt * 16) * 64 + kk * 32], 64, lane);
                oacc[dt] = wmma_f16(pa, vb, oacc[dt]);
            }
        }
    }

    // normalize and scatter to (T,B,E) f16 activation matrix
    for (int dt = 0; dt < 4; ++dt) {
        for (int i = 0; i < 8; ++i) {
            int t = t0 + i + 8 * g;
            int d = dt * 16 + ln;
            float v = oacc[dt][i] / l_i[i];
            int r = t * 4 + b;
            int e = h * 64 + d;
            A2[(size_t)r * EMB + e] = (_Float16)v;
        }
    }
}

// ---------------- launch ----------------
extern "C" void kernel_launch(void* const* d_in, const int* in_sizes, int n_in,
                              void* d_out, int out_size, void* d_ws, size_t ws_size,
                              hipStream_t stream) {
    (void)in_sizes; (void)n_in; (void)out_size; (void)ws_size;
    const float* query   = (const float*)d_in[0];
    const float* q_w     = (const float*)d_in[1];
    const float* q_b     = (const float*)d_in[2];
    const float* k_w     = (const float*)d_in[3];
    const float* k_b     = (const float*)d_in[4];
    const float* v_w     = (const float*)d_in[5];
    const float* v_b     = (const float*)d_in[6];
    const float* out_w   = (const float*)d_in[7];
    const float* out_b   = (const float*)d_in[8];
    const float* rel_emb = (const float*)d_in[9];
    const float* grep_w  = (const float*)d_in[10];
    const float* grep_b  = (const float*)d_in[11];
    const float* grep_a  = (const float*)d_in[12];
    const float* bias_k  = (const float*)d_in[13];
    const float* bias_v  = (const float*)d_in[14];

    char* ws = (char*)d_ws;
    size_t off = 0;
    auto alloc = [&](size_t bytes) {
        char* p = ws + off;
        off += (bytes + 255) & ~(size_t)255;
        return p;
    };
    _Float16* A    = (_Float16*)alloc((size_t)ROWS * EMB * 2);
    _Float16* Wqt  = (_Float16*)alloc((size_t)EMB * EMB * 2);
    _Float16* Wkt  = (_Float16*)alloc((size_t)EMB * EMB * 2);
    _Float16* Wvt  = (_Float16*)alloc((size_t)EMB * EMB * 2);
    _Float16* Wot  = (_Float16*)alloc((size_t)EMB * EMB * 2);
    _Float16* Qh   = (_Float16*)alloc((size_t)NB * TSEQ * HD * 2);
    _Float16* Kh   = (_Float16*)alloc((size_t)NB * SPAD * HD * 2);
    _Float16* Vtb  = (_Float16*)alloc((size_t)NB * HD * SPAD * 2);
    _Float16* A2   = (_Float16*)alloc((size_t)ROWS * EMB * 2);
    float*    gb   = (float*)alloc((size_t)NB * TSEQ * 4);
    float*    gws  = (float*)alloc(65 * 4);
    _Float16* pbh  = (_Float16*)alloc((size_t)NH * PBW * 2);

    k_cvt_f16<<<(ROWS * EMB / 4) / 256, 256, 0, stream>>>(query, A, ROWS * EMB);
    k_transpose_w<<<(EMB * EMB) / 256, 256, 0, stream>>>(q_w, Wqt);
    k_transpose_w<<<(EMB * EMB) / 256, 256, 0, stream>>>(k_w, Wkt);
    k_transpose_w<<<(EMB * EMB) / 256, 256, 0, stream>>>(v_w, Wvt);
    k_transpose_w<<<(EMB * EMB) / 256, 256, 0, stream>>>(out_w, Wot);
    k_pbh<<<(NH * PBW + 255) / 256, 256, 0, stream>>>(rel_emb, pbh);
    k_gwsum<<<1, 64, 0, stream>>>(grep_w, grep_b, gws);
    k_biasrows<<<(NB * HD * 64) / 256, 256, 0, stream>>>(bias_k, bias_v, Kh, Vtb);

    dim3 gg(ROWS / 128, EMB / 128);
    k_gemm_wmma<0><<<gg, 256, 0, stream>>>(A, Wqt, q_b, Qh);
    k_gemm_wmma<1><<<gg, 256, 0, stream>>>(A, Wkt, k_b, Kh);
    k_gemm_wmma<2><<<gg, 256, 0, stream>>>(A, Wvt, v_b, Vtb);

    k_gate<<<(NB * TSEQ) / 256, 256, 0, stream>>>(Qh, gws, grep_a, gb);

    dim3 ag(NB, TSEQ / 128);
    k_attn<<<ag, 256, 0, stream>>>(Qh, Kh, Vtb, gb, pbh, A2);

    k_gemm_wmma<3><<<gg, 256, 0, stream>>>(A2, Wot, out_b, d_out);
}
